// CopyGenerator_68152541053242
// MI455X (gfx1250) — matrix operand order repaired and checked by reading, not compiled, not run on hardware
//
#include <hip/hip_runtime.h>
#include <hip/hip_bf16.h>
#include <cstddef>

// Problem constants (match reference)
#define NT 128
#define NS 128
#define BSZ 8
#define DIM 512
#define VOC 32000

// ---------- types ----------
typedef __attribute__((ext_vector_type(8)))  __bf16 v8bf;
typedef __attribute__((ext_vector_type(16))) __bf16 v16bf;
typedef __attribute__((ext_vector_type(8)))  float  v8f;

union BF16Frag { v16bf v; v8bf h[2]; };

struct alignas(16) F4 { float x, y, z, w; };
struct alignas(8)  U4 { unsigned short a, b, c, d; };

__device__ __forceinline__ unsigned short f32_to_bf16_rne(float x) {
    unsigned u = __float_as_uint(x);
    u += 0x7FFFu + ((u >> 16) & 1u);   // round-to-nearest-even
    return (unsigned short)(u >> 16);
}

// ---------- f32 -> bf16 conversion (x4 vectorized) ----------
__global__ void k_cvt_bf16(const F4* __restrict__ in, U4* __restrict__ out, int n4) {
    int i = blockIdx.x * blockDim.x + threadIdx.x;
    if (i < n4) {
        F4 f = in[i];
        U4 o;
        o.a = f32_to_bf16_rne(f.x);
        o.b = f32_to_bf16_rne(f.y);
        o.c = f32_to_bf16_rne(f.z);
        o.d = f32_to_bf16_rne(f.w);
        out[i] = o;
    }
}

// ---------- w_eff[d] = sum_j copy_w[j] * f_w[j,d] ; bias_eff = copy_w.f_b + copy_b ----------
__global__ __launch_bounds__(DIM) void k_weff(
    const float* __restrict__ f_w, const float* __restrict__ f_b,
    const float* __restrict__ copy_w, const float* __restrict__ copy_b,
    float* __restrict__ w_eff, float* __restrict__ bias_eff)
{
    int d = threadIdx.x;
    float s = 0.f;
    for (int j = 0; j < DIM; ++j) s += copy_w[j] * f_w[(size_t)j * DIM + d];
    w_eff[d] = s;
    if (d == 0) {
        float bb = copy_b[0];
        for (int j = 0; j < DIM; ++j) bb += copy_w[j] * f_b[j];
        bias_eff[0] = bb;
    }
}

// ---------- WMMA GEMM: C[m,n] = sum_k A[m,k]*B[n,k] (+ bias[n]) ----------
// A: (M,K) row-major bf16, B: (N,K) row-major bf16, C: (M,N) f32.
// WG = 256 threads = 8 waves, tile 128(M) x 128(N); wave tile 32(M) x 64(N)
// = 2x4 WMMA 16x16 tiles; K stepped by 32 (v_wmma_f32_16x16x32_bf16).
// Fragment layouts per CDNA5 ISA 7.12.2 (wave32).
__global__ __launch_bounds__(256) void k_gemm_bf16(
    const __bf16* __restrict__ A, const __bf16* __restrict__ B,
    const float* __restrict__ bias, float* __restrict__ C,
    int M, int N, int K)
{
    const int lane  = threadIdx.x & 31;
    const int wave  = threadIdx.x >> 5;   // 0..7
    const int waveM = wave >> 1;          // 0..3
    const int waveN = wave & 1;           // 0..1
    const int mBase = blockIdx.y * 128 + waveM * 32;
    const int nBase = blockIdx.x * 128 + waveN * 64;
    const int lhalf = lane >> 4;          // 0 | 1
    const int lmod  = lane & 15;

    v8f zero = {};
    v8f acc[2][4];
#pragma unroll
    for (int i = 0; i < 2; ++i)
#pragma unroll
        for (int j = 0; j < 4; ++j) acc[i][j] = zero;

    const int aK = lhalf << 3;   // A: hi half-lanes hold K+8..15 / K+24..31
    const int bK = lhalf << 3;   // B: hi half-lanes hold K+8..15 / K+24..31

    for (int k0 = 0; k0 < K; k0 += 32) {
        v16bf afrag[2];
#pragma unroll
        for (int i = 0; i < 2; ++i) {
            const __bf16* ap = A + (size_t)(mBase + i * 16 + lmod) * K + k0 + aK;
            BF16Frag f;
            f.h[0] = *(const v8bf*)(ap);        // K 0..7   (lo) / 8..15  (hi)
            f.h[1] = *(const v8bf*)(ap + 16);   // K 16..23 (lo) / 24..31 (hi)
            afrag[i] = f.v;
        }
        v16bf bfrag[4];
#pragma unroll
        for (int j = 0; j < 4; ++j) {
            const __bf16* bp = B + (size_t)(nBase + j * 16 + lmod) * K + k0 + bK;
            BF16Frag f;
            f.h[0] = *(const v8bf*)(bp);        // K 0..7   (lo) / 8..15  (hi)
            f.h[1] = *(const v8bf*)(bp + 16);   // K 16..23 (lo) / 24..31 (hi)
            bfrag[j] = f.v;
        }
#pragma unroll
        for (int i = 0; i < 2; ++i)
#pragma unroll
            for (int j = 0; j < 4; ++j)
                acc[i][j] = __builtin_amdgcn_wmma_f32_16x16x32_bf16(
                    false, afrag[i], false, bfrag[j], (short)0, acc[i][j], false, false);
    }

    // C/D layout: VGPR r -> lanes 0-15: M=r, N=lane ; lanes 16-31: M=8+r, N=lane-16
#pragma unroll
    for (int i = 0; i < 2; ++i) {
        const int rowBase = mBase + i * 16 + (lhalf << 3);
#pragma unroll
        for (int j = 0; j < 4; ++j) {
            const int col = nBase + j * 16 + lmod;
            const float bv = bias ? bias[col] : 0.f;
#pragma unroll
            for (int r = 0; r < 8; ++r)
                C[(size_t)(rowBase + r) * N + col] = acc[i][j][r] + bv;
        }
    }
}

// ---------- kw[r] = dot(k[r,:], w_eff) ----------
__global__ __launch_bounds__(128) void k_rowdot(
    const float* __restrict__ kf, const float* __restrict__ w, float* __restrict__ out)
{
    __shared__ float red[128];
    const int r = blockIdx.x, tid = threadIdx.x;
    float s = 0.f;
    for (int d = tid; d < DIM; d += 128) s += kf[(size_t)r * DIM + d] * w[d];
    red[tid] = s; __syncthreads();
    for (int off = 64; off > 0; off >>= 1) {
        if (tid < off) red[tid] += red[tid + off];
        __syncthreads();
    }
    if (tid == 0) out[r] = red[0];
}

// ---------- attention logits + softmax + copy-gate, one WG per (t,b) ----------
__global__ __launch_bounds__(128) void k_attn(
    const float* __restrict__ q, const float* __restrict__ kf,
    const float* __restrict__ kw, const float* __restrict__ bias_eff,
    float* __restrict__ attn, float* __restrict__ a_copy)
{
    const int r = blockIdx.x;           // r = t*BSZ + b
    const int t = r / BSZ, b = r % BSZ;
    const int s = threadIdx.x;          // 0..127 (source position)
    __shared__ float qrow[DIM];
    __shared__ float red[128];
    for (int d = s; d < DIM; d += 128) qrow[d] = q[(size_t)r * DIM + d];
    __syncthreads();

    const float* krow = kf + (size_t)(s * BSZ + b) * DIM;
    float logit = 0.f;
    for (int d = 0; d < DIM; ++d) logit += qrow[d] * krow[d];
    logit *= 0.04419417382415922f;      // 1/sqrt(512)

    red[s] = logit; __syncthreads();
    for (int off = 64; off > 0; off >>= 1) {
        if (s < off) red[s] = fmaxf(red[s], red[s + off]);
        __syncthreads();
    }
    const float m = red[0]; __syncthreads();
    const float e = __expf(logit - m);
    red[s] = e; __syncthreads();
    for (int off = 64; off > 0; off >>= 1) {
        if (s < off) red[s] += red[s + off];
        __syncthreads();
    }
    const float a = e / red[0];
    attn[((size_t)b * NT + t) * NS + s] = a;
    __syncthreads();

    red[s] = a * kw[s * BSZ + b]; __syncthreads();
    for (int off = 64; off > 0; off >>= 1) {
        if (s < off) red[s] += red[s + off];
        __syncthreads();
    }
    if (s == 0) {
        const float lg = red[0] + bias_eff[0];
        a_copy[r] = 1.f / (1.f + __expf(-lg));
    }
}

// ---------- per-row max & sum(exp) over vocab (rows live in L2) ----------
__global__ __launch_bounds__(256) void k_rowstats(
    const float* __restrict__ gs, float* __restrict__ rmax, float* __restrict__ rz)
{
    __shared__ float red[256];
    const int r = blockIdx.x, tid = threadIdx.x;
    const float* row = gs + (size_t)r * VOC;
    float m = -3.4e38f;
    for (int v = tid; v < VOC; v += 256) m = fmaxf(m, row[v]);
    red[tid] = m; __syncthreads();
    for (int off = 128; off > 0; off >>= 1) {
        if (tid < off) red[tid] = fmaxf(red[tid], red[tid + off]);
        __syncthreads();
    }
    m = red[0]; __syncthreads();
    float s = 0.f;
    for (int v = tid; v < VOC; v += 256) s += __expf(row[v] - m);
    red[tid] = s; __syncthreads();
    for (int off = 128; off > 0; off >>= 1) {
        if (tid < off) red[tid] += red[tid + off];
        __syncthreads();
    }
    if (tid == 0) { rmax[r] = m; rz[r] = red[0]; }
}

// ---------- in-place: out = (1-a) * softmax(out) ----------
__global__ __launch_bounds__(256) void k_genp(
    float* __restrict__ out, const float* __restrict__ rmax,
    const float* __restrict__ rz, const float* __restrict__ acopy)
{
    const int r = blockIdx.y;
    const int v = blockIdx.x * 256 + threadIdx.x;   // 125 * 256 = 32000
    const size_t idx = (size_t)r * VOC + v;
    const float g = __expf(out[idx] - rmax[r]) / rz[r];
    out[idx] = (1.f - acopy[r]) * g;
}

// ---------- scatter a*attn into vocab positions ----------
__global__ __launch_bounds__(128) void k_scatter(
    float* __restrict__ out, const float* __restrict__ attn,
    const int* __restrict__ src, const float* __restrict__ acopy)
{
    const int r = blockIdx.x;           // t*BSZ + b
    const int t = r / BSZ, b = r % BSZ;
    const int s = threadIdx.x;
    const float w = acopy[r] * attn[((size_t)b * NT + t) * NS + s];
    const int vv = src[(size_t)s * BSZ + b];
    atomicAdd(out + (size_t)r * VOC + vv, w);
}

// ---------- final elementwise log ----------
__global__ __launch_bounds__(256) void k_log(float* __restrict__ out, size_t n) {
    const size_t i = (size_t)blockIdx.x * 256 + threadIdx.x;
    if (i < n) out[i] = __logf(out[i]);
}

// ---------------------------------------------------------------------------
extern "C" void kernel_launch(void* const* d_in, const int* in_sizes, int n_in,
                              void* d_out, int out_size, void* d_ws, size_t ws_size,
                              hipStream_t stream) {
    const float* htgt   = (const float*)d_in[0];
    const float* hsrc   = (const float*)d_in[1];
    const int*   src    = (const int*)  d_in[2];
    const float* emb    = (const float*)d_in[3];
    const float* q_w    = (const float*)d_in[4];
    const float* q_b    = (const float*)d_in[5];
    const float* f_w    = (const float*)d_in[6];
    const float* f_b    = (const float*)d_in[7];
    const float* copy_w = (const float*)d_in[8];
    const float* copy_b = (const float*)d_in[9];
    float* out = (float*)d_out;

    // workspace carve-out (~41 MB total; 256B aligned slabs)
    char* ws = (char*)d_ws;
    size_t off = 0;
    auto take = [&](size_t bytes) -> char* {
        char* p = ws + off;
        off += (bytes + 255) & ~(size_t)255;
        return p;
    };
    unsigned short* emb_bf  = (unsigned short*)take((size_t)VOC * DIM * 2);
    unsigned short* htgt_bf = (unsigned short*)take((size_t)NT * BSZ * DIM * 2);
    unsigned short* hsrc_bf = (unsigned short*)take((size_t)NS * BSZ * DIM * 2);
    unsigned short* qw_bf   = (unsigned short*)take((size_t)DIM * DIM * 2);
    float* qf    = (float*)take((size_t)NT * BSZ * DIM * 4);
    float* kf    = (float*)take((size_t)NS * BSZ * DIM * 4);
    float* attn  = (float*)take((size_t)BSZ * NT * NS * 4);
    float* kw    = (float*)take((size_t)NS * BSZ * 4);
    float* weff  = (float*)take((size_t)DIM * 4);
    float* beff  = (float*)take(256);
    float* acopy = (float*)take((size_t)NT * BSZ * 4);
    float* rmax  = (float*)take((size_t)NT * BSZ * 4);
    float* rz    = (float*)take((size_t)NT * BSZ * 4);

    const int M = NT * BSZ;   // 1024 fused (t,b) rows

    // 1) downconvert operands to bf16
    {
        int n4 = VOC * DIM / 4;
        k_cvt_bf16<<<(n4 + 255) / 256, 256, 0, stream>>>((const F4*)emb, (U4*)emb_bf, n4);
        n4 = M * DIM / 4;
        k_cvt_bf16<<<(n4 + 255) / 256, 256, 0, stream>>>((const F4*)htgt, (U4*)htgt_bf, n4);
        k_cvt_bf16<<<(n4 + 255) / 256, 256, 0, stream>>>((const F4*)hsrc, (U4*)hsrc_bf, n4);
        n4 = DIM * DIM / 4;
        k_cvt_bf16<<<(n4 + 255) / 256, 256, 0, stream>>>((const F4*)q_w, (U4*)qw_bf, n4);
    }

    // 2) collapsed copy-gate weights: w_eff = copy_w @ f_w, bias_eff
    k_weff<<<1, DIM, 0, stream>>>(f_w, f_b, copy_w, copy_b, weff, beff);

    // 3) q/k projections via WMMA (B = q_w is (N,K) row-major, same as needed)
    {
        dim3 grid(DIM / 128, M / 128);
        k_gemm_bf16<<<grid, 256, 0, stream>>>((const __bf16*)htgt_bf, (const __bf16*)qw_bf,
                                              q_b, qf, M, DIM, DIM);
        k_gemm_bf16<<<grid, 256, 0, stream>>>((const __bf16*)hsrc_bf, (const __bf16*)qw_bf,
                                              q_b, kf, M, DIM, DIM);
    }

    // 4) kw[s,b] = k_row . w_eff  (v == k in the reference)
    k_rowdot<<<NS * BSZ, 128, 0, stream>>>(kf, weff, kw);

    // 5) attention softmax + fused copy gate
    k_attn<<<M, 128, 0, stream>>>(qf, kf, kw, beff, attn, acopy);

    // 6) gen logits straight into d_out: (1024 x 32000 x 512) WMMA GEMM
    {
        dim3 grid(VOC / 128, M / 128);   // 250 x 8
        k_gemm_bf16<<<grid, 256, 0, stream>>>((const __bf16*)htgt_bf, (const __bf16*)emb_bf,
                                              nullptr, out, M, VOC, DIM);
    }

    // 7) softmax over vocab (rows L2-resident), mix with copy distribution, log
    k_rowstats<<<M, 256, 0, stream>>>(out, rmax, rz);
    k_genp<<<dim3(VOC / 256, M), 256, 0, stream>>>(out, rmax, rz, acopy);
    k_scatter<<<M, 128, 0, stream>>>(out, attn, src, acopy);
    {
        size_t n = (size_t)M * VOC;
        k_log<<<(unsigned)((n + 255) / 256), 256, 0, stream>>>(out, n);
    }
}
